// MoELayer_86406152061644
// MI455X (gfx1250) — compile-verified, matrix operand order checked by reading, not run on hardware
//
#include <hip/hip_runtime.h>
#include <hip/hip_bf16.h>
#include <math.h>

#define T_TOK 4096
#define D_DIM 1024
#define F_DIM 4096
#define N_EXP 8
#define M_TILE 32

typedef __attribute__((ext_vector_type(16))) __bf16         v16bf;
typedef __attribute__((ext_vector_type(8)))  float          v8f;
typedef __attribute__((ext_vector_type(4)))  unsigned int   u32x4;
typedef __attribute__((ext_vector_type(4)))  float          f32x4;
typedef __attribute__((ext_vector_type(4)))  unsigned short u16x4;

// float -> bf16, round-to-nearest-even
static __device__ __forceinline__ unsigned short f2bf(float f) {
    unsigned u = __float_as_uint(f);
    unsigned r = 0x7FFFu + ((u >> 16) & 1u);
    return (unsigned short)((u + r) >> 16);
}

// A/B fragment from bf16 memory (LDS or global): 8 halves at p (K=koff..koff+7),
// 8 halves at p+16 elements (K=16+koff..23+koff). Two 16B loads, no VALU.
static __device__ __forceinline__ v16bf load_frag(const unsigned short* p) {
    union { v16bf v; u32x4 q[2]; } u;
    u.q[0] = *(const u32x4*)(p);
    u.q[1] = *(const u32x4*)(p + 16);
    return u.v;
}

// ---------------------------------------------------------------------------
// Kernel 0: zero the output (we scatter-add into it)
// ---------------------------------------------------------------------------
__global__ void zero_kernel(float* __restrict__ p, int n) {
    int i = blockIdx.x * blockDim.x + threadIdx.x;
    if (i < n) p[i] = 0.0f;
}

// ---------------------------------------------------------------------------
// Kernel 0b: bulk f32 -> bf16 conversion (4 elems/thread, b128 in / b64 out)
// ---------------------------------------------------------------------------
__global__ __launch_bounds__(256) void cvt_bf16_kernel(const float* __restrict__ src,
                                                       unsigned short* __restrict__ dst,
                                                       int n4) {
    int i = blockIdx.x * blockDim.x + threadIdx.x;
    if (i < n4) {
        f32x4 v = *(const f32x4*)(src + (size_t)i * 4);
        u16x4 o;
        o.x = f2bf(v.x); o.y = f2bf(v.y); o.z = f2bf(v.z); o.w = f2bf(v.w);
        *(u16x4*)(dst + (size_t)i * 4) = o;
    }
}

// ---------------------------------------------------------------------------
// Kernel 1: router — one wave per token. 8 logits, top-2, renormalized softmax.
// ---------------------------------------------------------------------------
__global__ __launch_bounds__(256) void router_kernel(const float* __restrict__ x,
                                                     const float* __restrict__ Wr,
                                                     int* __restrict__ top_idx,
                                                     float* __restrict__ top_w) {
    int wave = (blockIdx.x * blockDim.x + threadIdx.x) >> 5;
    int lane = threadIdx.x & 31;
    if (wave >= T_TOK) return;
    const float* xr = x + (size_t)wave * D_DIM;

    float p[N_EXP];
#pragma unroll
    for (int e = 0; e < N_EXP; e++) p[e] = 0.0f;

    for (int k = lane; k < D_DIM; k += 32) {
        float xv = xr[k];
#pragma unroll
        for (int e = 0; e < N_EXP; e++) p[e] += xv * Wr[e * D_DIM + k];
    }
#pragma unroll
    for (int e = 0; e < N_EXP; e++)
        for (int off = 16; off > 0; off >>= 1) p[e] += __shfl_xor(p[e], off, 32);

    if (lane == 0) {
        int i0 = 0; float v0 = p[0];
#pragma unroll
        for (int e = 1; e < N_EXP; e++) if (p[e] > v0) { v0 = p[e]; i0 = e; }
        int i1 = (i0 == 0) ? 1 : 0; float v1 = p[i1];
#pragma unroll
        for (int e = 0; e < N_EXP; e++)
            if (e != i0 && p[e] > v1) { v1 = p[e]; i1 = e; }
        float e1 = __expf(v1 - v0);               // renorm top-2 == 2-way softmax
        float s  = 1.0f + e1;
        top_idx[wave * 2 + 0] = i0;
        top_idx[wave * 2 + 1] = i1;
        top_w[wave * 2 + 0]   = 1.0f / s;
        top_w[wave * 2 + 1]   = e1 / s;
    }
}

// ---------------------------------------------------------------------------
// Kernel 2: deterministic per-expert compaction (block scan, token order)
// ---------------------------------------------------------------------------
__global__ __launch_bounds__(256) void compact_kernel(const int* __restrict__ top_idx,
                                                      const float* __restrict__ top_w,
                                                      int* __restrict__ token_list,
                                                      float* __restrict__ gate,
                                                      int* __restrict__ counts) {
    const int e = blockIdx.x;
    const int tid = threadIdx.x, lane = tid & 31, wv = tid >> 5;
    __shared__ unsigned wcnt[8];
    __shared__ unsigned base;
    if (tid == 0) base = 0;
    __syncthreads();

    for (int c0 = 0; c0 < T_TOK; c0 += 256) {
        int t = c0 + tid;
        int i0 = top_idx[t * 2 + 0], i1 = top_idx[t * 2 + 1];
        bool m0 = (i0 == e), m1 = (i1 == e);
        bool match = m0 || m1;
        float w = m0 ? top_w[t * 2 + 0] : top_w[t * 2 + 1];

        unsigned bm = (unsigned)__ballot(match);
        unsigned prefix = __popc(bm & ((1u << lane) - 1u));
        if (lane == 0) wcnt[wv] = __popc(bm);
        __syncthreads();

        unsigned woff = 0, ctot = 0;
#pragma unroll
        for (int i = 0; i < 8; i++) { unsigned v = wcnt[i]; if (i < wv) woff += v; ctot += v; }
        unsigned mybase = base;
        __syncthreads();

        if (match) {
            unsigned pos = mybase + woff + prefix;
            token_list[e * T_TOK + pos] = t;
            gate[e * T_TOK + pos]       = w;
        }
        if (tid == 0) base += ctot;
        __syncthreads();
    }
    if (tid == 0) counts[e] = base;
}

// ---------------------------------------------------------------------------
// Kernel 3: fused expert FFN (WMMA bf16, f32 accumulate), M = 32 tokens / WG
//   grid = (128 token-tiles, 8 experts), block = 256 (8 waves)
//   F in chunks of 256: GEMM1 -> bias+GELU -> h(LDS) -> GEMM2 (y in regs)
// ---------------------------------------------------------------------------
__global__ __launch_bounds__(256) void moe_ffn_kernel(
        const unsigned short* __restrict__ xbf,   // [T, D] bf16
        const unsigned short* __restrict__ W1bf,  // [E, F, D] bf16
        const float* __restrict__ b1,
        const unsigned short* __restrict__ W2bf,  // [E, D, F] bf16
        const float* __restrict__ b2,
        const int* __restrict__ token_list, const float* __restrict__ gate,
        const int* __restrict__ counts, float* __restrict__ out) {
    const int e    = blockIdx.y;
    const int row0 = blockIdx.x * M_TILE;
    const int count = counts[e];
    if (row0 >= count) return;                       // uniform: whole WG exits

    __shared__ unsigned short x_lds[M_TILE * D_DIM]; // 64 KB bf16 token tile
    __shared__ unsigned short h_lds[M_TILE * 256];   // 16 KB bf16 activations
    __shared__ int   s_tok[M_TILE];
    __shared__ float s_gate[M_TILE];

    const int tid  = threadIdx.x;
    const int lane = tid & 31;
    const int wv   = tid >> 5;
    const int nn   = lane & 15;                      // N (or M) index within tile
    const int koff = (lane >> 4) << 3;               // K half select: 0 or 8
    const int mhi  = (lane >> 4) << 3;               // C/D row offset: 0 or 8

    if (tid < M_TILE) {
        int idx = row0 + tid;
        int tok = 0; float g = 0.0f;
        if (idx < count) { tok = token_list[e * T_TOK + idx]; g = gate[e * T_TOK + idx]; }
        s_tok[tid]  = tok;
        s_gate[tid] = g;
    }
    __syncthreads();

    // gather 32 token rows of bf16 x into LDS, 8 bytes per copy (no conversion)
#pragma unroll
    for (int i = 0; i < (M_TILE * D_DIM / 4) / 256; i++) {   // 32 iters
        int g4 = i * 256 + tid;                  // group of 4 halves
        int m  = g4 >> 8;                        // 256 groups per row
        int k4 = g4 & 255;
        *(u16x4*)(x_lds + m * D_DIM + k4 * 4) =
            *(const u16x4*)(xbf + (size_t)s_tok[m] * D_DIM + k4 * 4);
    }
    __syncthreads();

    const unsigned short* W1e = W1bf + (size_t)e * F_DIM * D_DIM;   // [F, D]
    const unsigned short* W2e = W2bf + (size_t)e * D_DIM * F_DIM;   // [D, F]

    v8f acc_y[2][8] = {};                        // [m-subtile][d-subtile]

#pragma unroll 1
    for (int fc = 0; fc < F_DIM; fc += 256) {
        // ---- GEMM1: h_tile[32 x 32] per wave, K = D = 1024 ----
        v8f acc1[2][2] = {};                     // [m-subtile][f-subtile]
        const int f_base = fc + wv * 32;
        const unsigned short* w1r0 = W1e + (size_t)(f_base +  0 + nn) * D_DIM;
        const unsigned short* w1r1 = W1e + (size_t)(f_base + 16 + nn) * D_DIM;
#pragma unroll 2
        for (int k0 = 0; k0 < D_DIM; k0 += 32) {
            v16bf a0 = load_frag(x_lds + (nn     ) * D_DIM + k0 + koff);
            v16bf a1 = load_frag(x_lds + (nn + 16) * D_DIM + k0 + koff);
            v16bf b0 = load_frag(w1r0 + k0 + koff);
            v16bf b1v = load_frag(w1r1 + k0 + koff);
            acc1[0][0] = __builtin_amdgcn_wmma_f32_16x16x32_bf16(false, a0, false, b0,
                                                                 (short)0, acc1[0][0], false, false);
            acc1[1][0] = __builtin_amdgcn_wmma_f32_16x16x32_bf16(false, a1, false, b0,
                                                                 (short)0, acc1[1][0], false, false);
            acc1[0][1] = __builtin_amdgcn_wmma_f32_16x16x32_bf16(false, a0, false, b1v,
                                                                 (short)0, acc1[0][1], false, false);
            acc1[1][1] = __builtin_amdgcn_wmma_f32_16x16x32_bf16(false, a1, false, b1v,
                                                                 (short)0, acc1[1][1], false, false);
        }

        __syncthreads();   // previous chunk's GEMM2 done reading h_lds
        // bias + exact GELU, store bf16 h tile
#pragma unroll
        for (int s = 0; s < 2; s++) {
            float bias = b1[e * F_DIM + f_base + s * 16 + nn];
            int col = wv * 32 + s * 16 + nn;
#pragma unroll
            for (int mi = 0; mi < 2; mi++) {
#pragma unroll
                for (int r = 0; r < 8; r++) {
                    int m = mi * 16 + r + mhi;
                    float v = acc1[mi][s][r] + bias;
                    float g = 0.5f * v * (1.0f + erff(v * 0.70710678118654752f));
                    h_lds[m * 256 + col] = f2bf(g);
                }
            }
        }
        __syncthreads();

        // ---- GEMM2: y += h_chunk @ W2[:, fc:fc+256]^T, N = 128 per wave ----
#pragma unroll 1
        for (int k2 = 0; k2 < 256; k2 += 32) {
            v16bf a0 = load_frag(h_lds + (nn     ) * 256 + k2 + koff);
            v16bf a1 = load_frag(h_lds + (nn + 16) * 256 + k2 + koff);
#pragma unroll
            for (int t = 0; t < 8; t++) {
                const unsigned short* w2r = W2e + (size_t)(wv * 128 + t * 16 + nn) * F_DIM
                                            + fc + k2 + koff;
                v16bf b = load_frag(w2r);
                acc_y[0][t] = __builtin_amdgcn_wmma_f32_16x16x32_bf16(false, a0, false, b,
                                                                      (short)0, acc_y[0][t], false, false);
                acc_y[1][t] = __builtin_amdgcn_wmma_f32_16x16x32_bf16(false, a1, false, b,
                                                                      (short)0, acc_y[1][t], false, false);
            }
        }
    }

    // epilogue: + b2, gate-weight, scatter-add (exactly 2 adds/elem, commutative)
#pragma unroll
    for (int t = 0; t < 8; t++) {
        int d = wv * 128 + t * 16 + nn;
        float bias = b2[e * D_DIM + d];
#pragma unroll
        for (int mi = 0; mi < 2; mi++) {
#pragma unroll
            for (int r = 0; r < 8; r++) {
                int m = mi * 16 + r + mhi;
                float val = s_gate[m] * (acc_y[mi][t][r] + bias);
                unsafeAtomicAdd(&out[(size_t)s_tok[m] * D_DIM + d], val);
            }
        }
    }
}

// ---------------------------------------------------------------------------
extern "C" void kernel_launch(void* const* d_in, const int* in_sizes, int n_in,
                              void* d_out, int out_size, void* d_ws, size_t ws_size,
                              hipStream_t stream) {
    const float* x  = (const float*)d_in[0];
    const float* Wr = (const float*)d_in[1];
    const float* W1 = (const float*)d_in[2];
    const float* b1 = (const float*)d_in[3];
    const float* W2 = (const float*)d_in[4];
    const float* b2 = (const float*)d_in[5];
    float* out = (float*)d_out;

    // ---- workspace layout ----
    char* ws = (char*)d_ws;
    const size_t W_ELEMS = (size_t)N_EXP * F_DIM * D_DIM;       // 33,554,432
    const size_t X_ELEMS = (size_t)T_TOK * D_DIM;               //  4,194,304
    int*            top_idx    = (int*)  (ws);                                  // 32 KB
    float*          top_w      = (float*)(ws + (size_t)T_TOK * 2 * 4);          // 32 KB
    int*            token_list = (int*)  (ws + (size_t)T_TOK * 4 * 4);          // 128 KB
    float*          gate       = (float*)(ws + (size_t)T_TOK * 4 * 4
                                             + (size_t)N_EXP * T_TOK * 4);      // 128 KB
    int*            counts     = (int*)  (ws + (size_t)T_TOK * 4 * 4
                                             + (size_t)N_EXP * T_TOK * 8);
    unsigned short* xbf        = (unsigned short*)(ws + 524288);
    unsigned short* W1bf       = xbf  + X_ELEMS;
    unsigned short* W2bf       = W1bf + W_ELEMS;
    // total ws use: 512 KB + 2*(4,194,304 + 2*33,554,432) B ~= 143 MB

    const int out_elems = T_TOK * D_DIM;
    zero_kernel<<<dim3((out_elems + 255) / 256), dim3(256), 0, stream>>>(out, out_elems);

    // one-shot bf16 conversions (puts full weight set ~134 MB inside the 192 MB L2)
    cvt_bf16_kernel<<<dim3((unsigned)(X_ELEMS / 4 / 256)), dim3(256), 0, stream>>>(x,  xbf,  (int)(X_ELEMS / 4));
    cvt_bf16_kernel<<<dim3((unsigned)(W_ELEMS / 4 / 256)), dim3(256), 0, stream>>>(W1, W1bf, (int)(W_ELEMS / 4));
    cvt_bf16_kernel<<<dim3((unsigned)(W_ELEMS / 4 / 256)), dim3(256), 0, stream>>>(W2, W2bf, (int)(W_ELEMS / 4));

    router_kernel<<<dim3(T_TOK / 8), dim3(256), 0, stream>>>(x, Wr, top_idx, top_w);
    compact_kernel<<<dim3(N_EXP), dim3(256), 0, stream>>>(top_idx, top_w, token_list, gate, counts);
    moe_ffn_kernel<<<dim3(T_TOK / M_TILE, N_EXP), dim3(256), 0, stream>>>(
        xbf, W1bf, b1, W2bf, b2, token_list, gate, counts, out);
}
